// Decoder_25795573579795
// MI455X (gfx1250) — matrix operand order repaired and checked by reading, not compiled, not run on hardware
//
#include <hip/hip_runtime.h>
#include <hip/hip_bf16.h>
#include <math.h>

// ---------------- compile-time dims (from reference) ----------------
#define TPB   256
#define NBLK  64
#define B_    32
#define TENC  512
#define E_    512
#define NMEL  80
#define TDEC  400
#define TT    (TDEC + 1)          // 401 scan steps
#define P_    256
#define A_    128
#define D_    1024
#define F_    32
#define KC    31
#define KPAD  96                  // prenet K padded 80 -> 96 (multiple of 32)
#define PREM  (TT * B_)           // 12832 prenet rows
#define HCW   (D_ + E_)           // 1536

#if defined(__has_builtin)
#if __has_builtin(__builtin_amdgcn_global_load_async_to_lds_b128) && \
    __has_builtin(__builtin_amdgcn_s_wait_asynccnt)
#define ASYNC_OK 1
#endif
#endif
#ifndef ASYNC_OK
#define ASYNC_OK 0
#endif

typedef int v4i __attribute__((vector_size(16)));

typedef __attribute__((ext_vector_type(16))) __bf16 v16bf;
typedef __attribute__((ext_vector_type(8)))  float  v8f;

union ABV { unsigned u[8]; v16bf v; };

__device__ __forceinline__ unsigned short f2bf(float x) {
  unsigned u = __builtin_bit_cast(unsigned, x);
  unsigned r = (u + 0x7FFFu + ((u >> 16) & 1u)) >> 16;   // round-to-nearest-even
  return (unsigned short)r;
}
__device__ __forceinline__ float bf2f(unsigned short b) {
  return __builtin_bit_cast(float, (unsigned)b << 16);
}
__device__ __forceinline__ float sigf(float x) { return 1.f / (1.f + __expf(-x)); }

// ---------------- grid-wide sense barrier (all NBLK blocks resident) --------
__device__ __forceinline__ void gbar(unsigned* bar) {
  __syncthreads();
  if (threadIdx.x == 0) {
    __threadfence();
    unsigned gen = __hip_atomic_load(&bar[1], __ATOMIC_RELAXED, __HIP_MEMORY_SCOPE_AGENT);
    unsigned arrived =
        __hip_atomic_fetch_add(&bar[0], 1u, __ATOMIC_ACQ_REL, __HIP_MEMORY_SCOPE_AGENT);
    if (arrived == (unsigned)(NBLK - 1)) {
      __hip_atomic_store(&bar[0], 0u, __ATOMIC_RELAXED, __HIP_MEMORY_SCOPE_AGENT);
      __hip_atomic_fetch_add(&bar[1], 1u, __ATOMIC_RELEASE, __HIP_MEMORY_SCOPE_AGENT);
    } else {
      while (__hip_atomic_load(&bar[1], __ATOMIC_ACQUIRE, __HIP_MEMORY_SCOPE_AGENT) == gen)
        __builtin_amdgcn_s_sleep(1);
    }
  }
  __syncthreads();
}

// ---------------- WMMA fragment loaders (CDNA5 wave32 layouts) --------------
// A 16x32 bf16 slab resident in LDS (row-major 16x32):
//   lane -> row M = lane&15; VGPR v holds K pair
//   k = (v<4?0:16) + (lane<16?0:8) + 2*(v&3)
__device__ __forceinline__ v16bf load_a_lds(const unsigned short* s, int lane) {
  ABV t;
  const unsigned short* row = s + (lane & 15) * 32 + ((lane & 16) ? 8 : 0);
#pragma unroll
  for (int v = 0; v < 8; ++v)
    t.u[v] = *(const unsigned*)(row + ((v & 4) ? 16 : 0) + 2 * (v & 3));
  return t.v;
}
// B 32x16 bf16 (row-major KxN source): lane -> col N = lane&15; VGPR v holds K pair
//   k = (v<4?0:16) + (lane<16?0:8) + 2*(v&3)
__device__ __forceinline__ v16bf load_b(const unsigned short* Bm, int ldb, int kb,
                                        int n0, int lane) {
  ABV t;
  const unsigned short* col = Bm + (size_t)kb * ldb + n0 + (lane & 15);
  const int h8 = (lane & 16) ? 8 : 0;
#pragma unroll
  for (int v = 0; v < 8; ++v) {
    const int k0 = ((v & 4) ? 16 : 0) + h8 + 2 * (v & 3);
    t.u[v] = (unsigned)col[(size_t)k0 * ldb] | ((unsigned)col[(size_t)(k0 + 1) * ldb] << 16);
  }
  return t.v;
}

// cooperative stage of one 16x32 bf16 A-slab (1KB) into LDS; threads 0..63
__device__ __forceinline__ void stageA(const unsigned short* A, int lda, int m0,
                                       int kb, unsigned short* dst, int tib) {
  if (tib < 64) {
    const int row = tib >> 2, c = (tib & 3) << 3;   // 4 x 16B per 64B row
    const unsigned short* src = A + (size_t)(m0 + row) * lda + kb + c;
#if ASYNC_OK
    __builtin_amdgcn_global_load_async_to_lds_b128(
        (v4i*)src, (v4i*)(dst + row * 32 + c), 0, 0);
#else
    const uint4 v = *(const uint4*)src;
    *(uint4*)(dst + row * 32 + c) = v;
#endif
  }
}

// ---- block-cooperative dual-K GEMM: C = act(A1*B1 + A2*B2 + bias) ----------
// A row-major MxK bf16 (staged via LDS, shared by the 8 waves of the block),
// B row-major KxN bf16 streamed from global, C row-major MxN f32/bf16.
// All N here are multiples of 128 so each group has exactly 8 full N-tiles.
__device__ void gemm_block(const unsigned short* A1, const unsigned short* B1, int K1,
                           const unsigned short* A2, const unsigned short* B2, int K2,
                           const float* bias, float* C, unsigned short* Cbf,
                           int M, int N, int act, unsigned short (*lA)[512]) {
  const int tib  = threadIdx.x;
  const int lane = tib & 31;
  const int wib  = tib >> 5;                 // wave in block: 0..7
  const int tilesN = N >> 4;
  const int gN     = tilesN >> 3;            // N-groups (8 tiles each)
  const int groups = (M >> 4) * gN;
  const int nkb    = (K1 + K2) >> 5;
  for (int g = blockIdx.x; g < groups; g += gridDim.x) {
    const int m0 = (g / gN) << 4;
    const int n0 = (((g % gN) << 3) + wib) << 4;
    v8f acc = {};
    stageA(A1, K1, m0, 0, lA[0], tib);
#if ASYNC_OK
    __builtin_amdgcn_s_wait_asynccnt(0);
#endif
    __syncthreads();
    for (int kbi = 0; kbi < nkb; ++kbi) {
      if (kbi + 1 < nkb) {                    // prefetch next slab (double buffer)
        const int kn = (kbi + 1) << 5;
        if (kn < K1) stageA(A1, K1, m0, kn, lA[(kbi + 1) & 1], tib);
        else         stageA(A2, K2, m0, kn - K1, lA[(kbi + 1) & 1], tib);
      }
      const int k = kbi << 5;
      const unsigned short* Bm; int kb, Kc;
      if (k < K1) { Bm = B1; kb = k; Kc = K1; } else { Bm = B2; kb = k - K1; Kc = K2; }
      if (kb + 32 < Kc) __builtin_prefetch(Bm + (size_t)(kb + 32) * N + n0, 0, 3);
      v16bf a = load_a_lds(lA[kbi & 1], lane);
      v16bf b = load_b(Bm, N, kb, n0, lane);
      acc = __builtin_amdgcn_wmma_f32_16x16x32_bf16(false, a, false, b,
                                                    (short)0, acc, false, false);
#if ASYNC_OK
      __builtin_amdgcn_s_wait_asynccnt(0);
#endif
      __syncthreads();
    }
    const int n  = n0 + (lane & 15);
    const float bv = bias ? bias[n] : 0.f;
#pragma unroll
    for (int r = 0; r < 8; ++r) {
      const int m = m0 + r + ((lane & 16) ? 8 : 0);
      float x = acc[r] + bv;
      if (act == 1) x = fmaxf(x, 0.f);
      if (C)   C[(size_t)m * N + n] = x;
      if (Cbf) Cbf[(size_t)m * N + n] = f2bf(x);
    }
  }
}

// ---------------- parameters -----------------------------------------------
struct KParams {
  // fp32 inputs
  const float *enc, *mel, *pb1, *pb2, *ab, *cw, *vvec, *db, *gw, *gb;
  const float *pw1, *pw2, *awx, *awh, *wq, *wmem, *wloc, *dwx, *dwh;
  // bf16 workspace
  unsigned short *bw_pre1, *bw_pre2, *bw_awx, *bw_awh, *bw_q, *bw_mem, *bw_loc,
      *bw_dwx, *bw_dwh, *enc_bf, *a_pre, *hid1_bf, *pre_bf, *xh_att, *convo,
      *atth_bf, *xh_dec, *dech_bf, *ctx_bf;
  // f32 workspace
  float *wenc, *z_att, *att_h, *att_c, *pq, *locW, *e, *aw, *awc,
      *ctx, *z_dec, *dec_h, *dec_c;
  unsigned* bar;
  float *out_mels, *out_gate;
};

// ---------------- persistent decoder kernel --------------------------------
__global__ __launch_bounds__(TPB) void tacotron_decoder(KParams p) {
  const int tid = blockIdx.x * blockDim.x + threadIdx.x;
  const int nth = gridDim.x * blockDim.x;             // 16384
  __shared__ unsigned short lA[2][512];               // double-buffered A slab
  __shared__ float sm[TPB];

  // ===== precompute phase 1: bf16 weight conversion + padded inputs + state 0
  for (int i = tid; i < KPAD * P_; i += nth) {        // prenet W1 padded 96x256
    int k = i / P_, n = i % P_;
    p.bw_pre1[i] = (k < NMEL) ? f2bf(p.pw1[k * P_ + n]) : (unsigned short)0;
  }
  for (int i = tid; i < P_ * P_; i += nth)        p.bw_pre2[i] = f2bf(p.pw2[i]);
  for (int i = tid; i < 768 * 512; i += nth)      p.bw_awx[i]  = f2bf(p.awx[i]);
  for (int i = tid; i < A_ * 512; i += nth)       p.bw_awh[i]  = f2bf(p.awh[i]);
  for (int i = tid; i < A_ * A_; i += nth)        p.bw_q[i]    = f2bf(p.wq[i]);
  for (int i = tid; i < E_ * A_; i += nth)        p.bw_mem[i]  = f2bf(p.wmem[i]);
  for (int i = tid; i < F_ * A_; i += nth)        p.bw_loc[i]  = f2bf(p.wloc[i]);
  for (int i = tid; i < 640 * 4096; i += nth)     p.bw_dwx[i]  = f2bf(p.dwx[i]);
  for (int i = tid; i < D_ * 4096; i += nth)      p.bw_dwh[i]  = f2bf(p.dwh[i]);
  for (int i = tid; i < B_ * TENC * E_; i += nth) p.enc_bf[i]  = f2bf(p.enc[i]);
  for (int i = tid; i < PREM * KPAD; i += nth) {      // teacher-forced mel, padded
    int r = i / KPAD, j = i % KPAD;
    int t = r / B_, b = r % B_;
    float v = 0.f;
    if (j < NMEL && t > 0) v = p.mel[(size_t)b * (NMEL * TDEC) + (t - 1) * NMEL + j];
    p.a_pre[i] = f2bf(v);
  }
  for (int i = tid; i < B_ * A_; i += nth) {
    p.att_h[i] = 0.f; p.att_c[i] = 0.f; p.atth_bf[i] = 0;
  }
  for (int i = tid; i < B_ * D_; i += nth) {
    p.dec_h[i] = 0.f; p.dec_c[i] = 0.f; p.dech_bf[i] = 0;
  }
  for (int i = tid; i < B_ * E_; i += nth) { p.ctx[i] = 0.f; p.ctx_bf[i] = 0; }
  for (int i = tid; i < B_ * TENC; i += nth) { p.aw[i] = 0.f; p.awc[i] = 0.f; }
  gbar(p.bar);

  // ===== precompute phase 2: prenet layer1 (12832x96 @ 96x256, relu)
  gemm_block(p.a_pre, p.bw_pre1, KPAD, nullptr, nullptr, 0, p.pb1,
             nullptr, p.hid1_bf, PREM, P_, 1, lA);
  gbar(p.bar);

  // ===== precompute phase 3: prenet layer2 + processed memory W_enc
  gemm_block(p.hid1_bf, p.bw_pre2, P_, nullptr, nullptr, 0, p.pb2,
             nullptr, p.pre_bf, PREM, P_, 1, lA);
  gemm_block(p.enc_bf, p.bw_mem, E_, nullptr, nullptr, 0, nullptr,
             p.wenc, nullptr, B_ * TENC, A_, 0, lA);
  gbar(p.bar);

  // ===== 401-step sequential scan ==========================================
  for (int t = 0; t < TT; ++t) {
    // --- phase A: assemble attention-LSTM input + location conv ------------
    for (int i = tid; i < B_ * 768; i += nth) {
      int b = i / 768, j = i % 768;
      p.xh_att[i] = (j < P_) ? p.pre_bf[(size_t)(t * B_ + b) * P_ + j]
                             : f2bf(p.ctx[b * E_ + (j - P_)]);
    }
    for (int i = tid; i < B_ * TENC; i += nth) {      // conv over [aw, awc]
      int b = i / TENC, tt = i % TENC;
      float wa[KC], wc[KC];                           // hoist 62-tap window
#pragma unroll
      for (int k = 0; k < KC; ++k) {
        int tp = tt + k - (KC / 2);
        bool ok = (tp >= 0) && (tp < TENC);
        wa[k] = ok ? p.aw[b * TENC + tp] : 0.f;
        wc[k] = ok ? p.awc[b * TENC + tp] : 0.f;
      }
      for (int f = 0; f < F_; ++f) {
        float acc = 0.f;
#pragma unroll
        for (int k = 0; k < KC; ++k)
          acc += wa[k] * p.cw[k * (2 * F_) + f] + wc[k] * p.cw[k * (2 * F_) + F_ + f];
        p.convo[(size_t)i * F_ + f] = f2bf(acc);
      }
    }
    gbar(p.bar);

    // --- attention LSTM pre-activations: z = xh@Wx + h@Wh + b --------------
    gemm_block(p.xh_att, p.bw_awx, 768, p.atth_bf, p.bw_awh, A_, p.ab,
               p.z_att, nullptr, B_, 4 * A_, 0, lA);
    gbar(p.bar);

    // --- attention LSTM pointwise ------------------------------------------
    for (int i = tid; i < B_ * A_; i += nth) {
      int b = i / A_, u = i % A_;
      const float* z = p.z_att + (size_t)b * (4 * A_);
      float ig = sigf(z[u]), fg = sigf(z[A_ + u]);
      float gg = tanhf(z[2 * A_ + u]), og = sigf(z[3 * A_ + u]);
      float c2 = fg * p.att_c[i] + ig * gg;
      float h2 = og * tanhf(c2);
      p.att_c[i] = c2; p.att_h[i] = h2; p.atth_bf[i] = f2bf(h2);
    }
    gbar(p.bar);

    // --- pq = att_h @ Wq   and   locW = conv_feat @ Wloc -------------------
    gemm_block(p.atth_bf, p.bw_q, A_, nullptr, nullptr, 0, nullptr,
               p.pq, nullptr, B_, A_, 0, lA);
    gemm_block(p.convo, p.bw_loc, F_, nullptr, nullptr, 0, nullptr,
               p.locW, nullptr, B_ * TENC, A_, 0, lA);
    gbar(p.bar);

    // --- attention energies e = tanh(pq + locW + Wenc) @ v -----------------
    for (int i = tid; i < B_ * TENC; i += nth) {
      const float* pqb = p.pq + (size_t)(i / TENC) * A_;
      const float* lw  = p.locW + (size_t)i * A_;
      const float* we  = p.wenc + (size_t)i * A_;
      float s = 0.f;
      for (int a = 0; a < A_; ++a) s += tanhf(pqb[a] + lw[a] + we[a]) * p.vvec[a];
      p.e[i] = s;
    }
    gbar(p.bar);

    // --- softmax over Tenc (one block per batch row) -----------------------
    if (blockIdx.x < B_) {
      int b = blockIdx.x;
      float e0 = p.e[b * TENC + threadIdx.x];
      float e1 = p.e[b * TENC + TPB + threadIdx.x];
      sm[threadIdx.x] = fmaxf(e0, e1); __syncthreads();
      for (int s = TPB / 2; s > 0; s >>= 1) {
        if (threadIdx.x < s) sm[threadIdx.x] = fmaxf(sm[threadIdx.x], sm[threadIdx.x + s]);
        __syncthreads();
      }
      float mx = sm[0]; __syncthreads();
      float x0 = __expf(e0 - mx), x1 = __expf(e1 - mx);
      sm[threadIdx.x] = x0 + x1; __syncthreads();
      for (int s = TPB / 2; s > 0; s >>= 1) {
        if (threadIdx.x < s) sm[threadIdx.x] += sm[threadIdx.x + s];
        __syncthreads();
      }
      float inv = 1.f / sm[0]; __syncthreads();
      float a0 = x0 * inv, a1 = x1 * inv;
      p.aw[b * TENC + threadIdx.x] = a0;
      p.aw[b * TENC + TPB + threadIdx.x] = a1;
      p.awc[b * TENC + threadIdx.x] += a0;
      p.awc[b * TENC + TPB + threadIdx.x] += a1;
    }
    gbar(p.bar);

    // --- context: ctx[b,e] = sum_t aw[b,t] * enc[b,t,e]  (bf16 encoder) ----
    for (int i = tid; i < B_ * E_; i += nth) {
      int b = i >> 9, ei = i & (E_ - 1);
      const float* awb = p.aw + b * TENC;
      const unsigned short* eb = p.enc_bf + (size_t)b * TENC * E_ + ei;
      float s = 0.f;
      for (int tt = 0; tt < TENC; ++tt) s += awb[tt] * bf2f(eb[(size_t)tt * E_]);
      p.ctx[i] = s; p.ctx_bf[i] = f2bf(s);
    }
    gbar(p.bar);

    // --- assemble decoder-LSTM input [att_h, ctx] --------------------------
    for (int i = tid; i < B_ * 640; i += nth) {
      int b = i / 640, j = i % 640;
      float v = (j < A_) ? p.att_h[b * A_ + j] : p.ctx[b * E_ + (j - A_)];
      p.xh_dec[i] = f2bf(v);
    }
    gbar(p.bar);

    // --- decoder LSTM pre-activations (the big one: 32x4096, K=640+1024) ---
    gemm_block(p.xh_dec, p.bw_dwx, 640, p.dech_bf, p.bw_dwh, D_, p.db,
               p.z_dec, nullptr, B_, 4 * D_, 0, lA);
    gbar(p.bar);

    // --- decoder LSTM pointwise --------------------------------------------
    for (int i = tid; i < B_ * D_; i += nth) {
      int b = i / D_, u = i % D_;
      const float* z = p.z_dec + (size_t)b * (4 * D_);
      float ig = sigf(z[u]), fg = sigf(z[D_ + u]);
      float gg = tanhf(z[2 * D_ + u]), og = sigf(z[3 * D_ + u]);
      float c2 = fg * p.dec_c[i] + ig * gg;
      float h2 = og * tanhf(c2);
      p.dec_c[i] = c2; p.dec_h[i] = h2; p.dech_bf[i] = f2bf(h2);
    }
    gbar(p.bar);

    // --- emit hc = [dec_h, ctx] and gate -----------------------------------
    for (int i = tid; i < B_ * HCW; i += nth) {
      int b = i / HCW, j = i % HCW;
      float v = (j < D_) ? p.dec_h[b * D_ + j] : p.ctx[b * E_ + (j - D_)];
      p.out_mels[(size_t)t * (B_ * HCW) + i] = v;
    }
    if (tid < B_) {
      float s = p.gb[0];
      for (int j = 0; j < HCW; ++j) {
        float hv = (j < D_) ? p.dec_h[tid * D_ + j] : p.ctx[tid * E_ + (j - D_)];
        s += hv * p.gw[j];
      }
      p.out_gate[t * B_ + tid] = s;
    }
    gbar(p.bar);
  }
}

// ---------------- host-side launch ------------------------------------------
extern "C" void kernel_launch(void* const* d_in, const int* in_sizes, int n_in,
                              void* d_out, int out_size, void* d_ws, size_t ws_size,
                              hipStream_t stream) {
  (void)in_sizes; (void)n_in; (void)out_size; (void)ws_size;
  KParams p;
  p.enc  = (const float*)d_in[0];  p.mel  = (const float*)d_in[1];
  p.pw1  = (const float*)d_in[2];  p.pb1  = (const float*)d_in[3];
  p.pw2  = (const float*)d_in[4];  p.pb2  = (const float*)d_in[5];
  p.awx  = (const float*)d_in[6];  p.awh  = (const float*)d_in[7];
  p.ab   = (const float*)d_in[8];  p.wq   = (const float*)d_in[9];
  p.wmem = (const float*)d_in[10]; p.cw   = (const float*)d_in[11];
  p.wloc = (const float*)d_in[12]; p.vvec = (const float*)d_in[13];
  p.dwx  = (const float*)d_in[14]; p.dwh  = (const float*)d_in[15];
  p.db   = (const float*)d_in[16]; p.gw   = (const float*)d_in[17];
  p.gb   = (const float*)d_in[18];

  char* w = (char*)d_ws;
  auto take = [&](size_t bytes) {
    char* r = w;
    w += (bytes + 255) & ~(size_t)255;
    return r;
  };
  p.bar     = (unsigned*)take(2 * sizeof(unsigned));
  // bf16 buffers
  p.bw_pre1 = (unsigned short*)take(2ull * KPAD * P_);
  p.bw_pre2 = (unsigned short*)take(2ull * P_ * P_);
  p.bw_awx  = (unsigned short*)take(2ull * 768 * 512);
  p.bw_awh  = (unsigned short*)take(2ull * A_ * 512);
  p.bw_q    = (unsigned short*)take(2ull * A_ * A_);
  p.bw_mem  = (unsigned short*)take(2ull * E_ * A_);
  p.bw_loc  = (unsigned short*)take(2ull * F_ * A_);
  p.bw_dwx  = (unsigned short*)take(2ull * 640 * 4096);
  p.bw_dwh  = (unsigned short*)take(2ull * D_ * 4096);
  p.enc_bf  = (unsigned short*)take(2ull * B_ * TENC * E_);
  p.a_pre   = (unsigned short*)take(2ull * PREM * KPAD);
  p.hid1_bf = (unsigned short*)take(2ull * PREM * P_);
  p.pre_bf  = (unsigned short*)take(2ull * PREM * P_);
  p.xh_att  = (unsigned short*)take(2ull * B_ * 768);
  p.convo   = (unsigned short*)take(2ull * B_ * TENC * F_);
  p.atth_bf = (unsigned short*)take(2ull * B_ * A_);
  p.xh_dec  = (unsigned short*)take(2ull * B_ * 640);
  p.dech_bf = (unsigned short*)take(2ull * B_ * D_);
  p.ctx_bf  = (unsigned short*)take(2ull * B_ * E_);
  // f32 buffers
  p.wenc  = (float*)take(4ull * B_ * TENC * A_);
  p.z_att = (float*)take(4ull * B_ * 4 * A_);
  p.att_h = (float*)take(4ull * B_ * A_);
  p.att_c = (float*)take(4ull * B_ * A_);
  p.pq    = (float*)take(4ull * B_ * A_);
  p.locW  = (float*)take(4ull * B_ * TENC * A_);
  p.e     = (float*)take(4ull * B_ * TENC);
  p.aw    = (float*)take(4ull * B_ * TENC);
  p.awc   = (float*)take(4ull * B_ * TENC);
  p.ctx   = (float*)take(4ull * B_ * E_);
  p.z_dec = (float*)take(4ull * B_ * 4 * D_);
  p.dec_h = (float*)take(4ull * B_ * D_);
  p.dec_c = (float*)take(4ull * B_ * D_);

  p.out_mels = (float*)d_out;
  p.out_gate = (float*)d_out + (size_t)TT * B_ * HCW;

  (void)hipMemsetAsync(p.bar, 0, 2 * sizeof(unsigned), stream);
  tacotron_decoder<<<NBLK, TPB, 0, stream>>>(p);
}
